// GCN_14405320310896
// MI455X (gfx1250) — matrix operand order repaired and checked by reading, not compiled, not run on hardware
//
#include <hip/hip_runtime.h>
#include <hip/hip_bf16.h>

typedef __attribute__((ext_vector_type(16))) _Float16 v16h;
typedef __attribute__((ext_vector_type(8)))  float    v8f;

#define N_NODES 100000
#define N_EDGES 600000
#define NODE_DIM 64
#define EDGE_DIM 32
#define HID 128
#define OUT 24

// ---------------------------------------------------------------------------
// WMMA fragment layouts (CDNA5 ISA §7.12.2, wave32):
//   A 16x32 f16 : lane l -> row l%16; elem i -> K=(i<8?i:i+8)+(l>=16?8:0)
//   B 32x16 f16 : lane l -> col l%16; elem i -> K=(i<8?i:i+8)+(l>=16?8:0)
//   C/D 16x16 f32: reg r -> M=r+(l>=16?8:0), N=l%16
// All global addressing below uses uniform SGPR base + 32-bit lane offset so
// the backend emits GVS-mode vmem ops (saddr + voffset, scale_offset) with no
// per-access 64-bit VALU address math.
// ---------------------------------------------------------------------------

// A-fragment: two contiguous 8-float runs -> 4x global_load_b128 + cvt.
__device__ __forceinline__ v16h load_a_frag_off(const float* __restrict__ base,
                                                int rowoff /*floats*/, int k0) {
    const int lane = threadIdx.x & 31;
    const int hi   = lane >> 4;
    const float4* p4 = (const float4*)base;
    const int o = (rowoff + k0 + (hi ? 8 : 0)) >> 2;   // float4 units
    const float4 q0 = p4[o], q1 = p4[o + 1], q2 = p4[o + 4], q3 = p4[o + 5];
    v16h a;
    a[0]  = (_Float16)q0.x; a[1]  = (_Float16)q0.y; a[2]  = (_Float16)q0.z; a[3]  = (_Float16)q0.w;
    a[4]  = (_Float16)q1.x; a[5]  = (_Float16)q1.y; a[6]  = (_Float16)q1.z; a[7]  = (_Float16)q1.w;
    a[8]  = (_Float16)q2.x; a[9]  = (_Float16)q2.y; a[10] = (_Float16)q2.z; a[11] = (_Float16)q2.w;
    a[12] = (_Float16)q3.x; a[13] = (_Float16)q3.y; a[14] = (_Float16)q3.z; a[15] = (_Float16)q3.w;
    return a;
}

// Same pattern from an LDS row pointer (ds_load_b128).
__device__ __forceinline__ v16h load_a_frag_lds(const float* rowp, int k0) {
    const int lane = threadIdx.x & 31;
    const int hi   = lane >> 4;
    const float4* p = (const float4*)(rowp + k0 + (hi ? 8 : 0));
    const float4 q0 = p[0], q1 = p[1], q2 = p[4], q3 = p[5];
    v16h a;
    a[0]  = (_Float16)q0.x; a[1]  = (_Float16)q0.y; a[2]  = (_Float16)q0.z; a[3]  = (_Float16)q0.w;
    a[4]  = (_Float16)q1.x; a[5]  = (_Float16)q1.y; a[6]  = (_Float16)q1.z; a[7]  = (_Float16)q1.w;
    a[8]  = (_Float16)q2.x; a[9]  = (_Float16)q2.y; a[10] = (_Float16)q2.z; a[11] = (_Float16)q2.w;
    a[12] = (_Float16)q3.x; a[13] = (_Float16)q3.y; a[14] = (_Float16)q3.z; a[15] = (_Float16)q3.w;
    return a;
}

// B-fragment from pre-packed f16 buffer: one 32B contiguous chunk per lane.
__device__ __forceinline__ v16h load_b_frag_packed(const _Float16* __restrict__ P, int frag) {
    const int lane = threadIdx.x & 31;
    return *(const v16h*)(P + (((size_t)frag * 32 + lane) << 4));
}

__device__ __forceinline__ v8f wmma_f16(v16h a, v16h b, v8f c) {
    return __builtin_amdgcn_wmma_f32_16x16x32_f16(false, a, false, b, (short)0, c,
                                                  false, false);
}

// ---------------------------------------------------------------------------
// Pre-pack a row-major f32 weight W[K x N] (ldw = N) into f16 fragments in the
// per-lane B layout. Fragment index = kc*tiles + t. Cols >= N zero-padded.
// ---------------------------------------------------------------------------
__global__ void pack_w_f16(const float* __restrict__ W, _Float16* __restrict__ P,
                           int K, int N, int tiles) {
    const int idx   = blockIdx.x * blockDim.x + threadIdx.x;
    const int total = (K / 32) * tiles * 512;
    if (idx >= total) return;
    const int i    = idx & 15;
    const int lane = (idx >> 4) & 31;
    const int frag = idx >> 9;
    const int kc   = frag / tiles;
    const int t    = frag - kc * tiles;
    const int k    = kc * 32 + ((lane >> 4) ? 8 : 0) + (i < 8 ? i : i + 8);
    const int col  = t * 16 + (lane & 15);
    P[idx] = (col < N) ? (_Float16)W[(size_t)k * N + col] : (_Float16)0.0f;
}

// ---------------------------------------------------------------------------
// GEMM: C[M x 128] = act(A[M x K] @ W[K x 128] + bias); optional dual store to
// C2 (seeds next layer's accumulator: z = h + agg with eps = 0).
// block = 256 (8 waves); wave w owns column tile w; blockIdx.x = row tile.
// ---------------------------------------------------------------------------
template <bool RELU>
__global__ void __launch_bounds__(256)
gemm_bias_act(const float* __restrict__ A, const _Float16* __restrict__ Wp,
              const float* __restrict__ bias, float* __restrict__ C,
              float* __restrict__ C2, int M, int K) {
    const int wave = threadIdx.x >> 5;
    const int lane = threadIdx.x & 31;
    const int hi   = lane >> 4;
    const int ln   = lane & 15;
    const int row0 = blockIdx.x * 16;
    if (row0 >= M) return;

    const int aoff = (row0 + ln) * K;
    v8f acc = {};
    for (int k0 = 0; k0 < K; k0 += 32) {
        v16h a = load_a_frag_off(A, aoff, k0);
        v16h b = load_b_frag_packed(Wp, (k0 >> 5) * 8 + wave);
        acc = wmma_f16(a, b, acc);
    }
    const int col  = wave * 16 + ln;
    const int coff = (row0 + (hi ? 8 : 0)) * HID + col;   // + r*128
    const float bv = bias[col];
#pragma unroll
    for (int r = 0; r < 8; ++r) {
        float v = acc[r] + bv;
        if (RELU) v = fmaxf(v, 0.0f);
        C[coff + r * HID] = v;
        if (C2) C2[coff + r * HID] = v;
    }
}

// ---------------------------------------------------------------------------
// Fused edge message + scatter:
//   e_emb = edge_attr @ ew + eb  (one WMMA per 16-edge x 16-col tile)
//   m     = relu(h[src] + e_emb);  agg[dst] += m  (native f32 atomics)
// block = 256 (8 waves); each wave handles 16 edges. All gather/scatter
// addressing: 32-bit row offsets + compile-time immediates.
// ---------------------------------------------------------------------------
__global__ void __launch_bounds__(256)
edge_msg_scatter(const float* __restrict__ edge_attr,
                 const int* __restrict__ src, const int* __restrict__ dst,
                 const _Float16* __restrict__ ewp, const float* __restrict__ eb,
                 const float* __restrict__ h, float* __restrict__ agg, int E) {
    const int wave = threadIdx.x >> 5;
    const int lane = threadIdx.x & 31;
    const int hi   = lane >> 4;
    const int ln   = lane & 15;
    const int e0   = (blockIdx.x * 8 + wave) * 16;
    if (e0 >= E) return;  // wave-uniform: EXEC stays all-ones for WMMA

    v16h a = load_a_frag_off(edge_attr, (e0 + ln) * EDGE_DIM, 0);

    // 32-bit row offsets (lane column folded in); accesses use +t*16 imm.
    int ho[8], ao[8];
#pragma unroll
    for (int r = 0; r < 8; ++r) {
        const int m = e0 + (hi ? 8 : 0) + r;
        ho[r] = src[m] * HID + ln;
        ao[r] = dst[m] * HID + ln;
    }

#pragma unroll
    for (int t = 0; t < 8; ++t) {
        v16h b  = load_b_frag_packed(ewp, t);
        v8f acc = {};
        acc = wmma_f16(a, b, acc);
        const float bv = eb[t * 16 + ln];
#pragma unroll
        for (int r = 0; r < 8; ++r) {
            const float v = fmaxf(acc[r] + bv + h[ho[r] + t * 16], 0.0f);
            __hip_atomic_fetch_add(&agg[ao[r] + t * 16], v,
                                   __ATOMIC_RELAXED, __HIP_MEMORY_SCOPE_AGENT);
        }
    }
}

// ---------------------------------------------------------------------------
// Fused edge classifier:
//   feat = [h[src] | h[dst] | edge_attr]  (K=288, never materialized)
//   hid  = relu(feat @ w1 + b1)           (staged in LDS)
//   out  = hid @ w2 + b2                  (N=24, padded tile)
// block = 128 (4 waves); each wave handles 16 edges.
// ---------------------------------------------------------------------------
__global__ void __launch_bounds__(128)
edge_classifier(const float* __restrict__ h,
                const int* __restrict__ src, const int* __restrict__ dst,
                const float* __restrict__ edge_attr,
                const _Float16* __restrict__ w1p, const float* __restrict__ b1,
                const _Float16* __restrict__ w2p, const float* __restrict__ b2,
                float* __restrict__ out, int E) {
    __shared__ float hid[4][16][132];
    const int wave = threadIdx.x >> 5;
    const int lane = threadIdx.x & 31;
    const int hi   = lane >> 4;
    const int ln   = lane & 15;
    const int e0   = (blockIdx.x * 4 + wave) * 16;

    const int afrow = e0 + ln;
    const int so = src[afrow] * HID;       // 32-bit gather row offsets
    const int dofs = dst[afrow] * HID;
    const int eo = afrow * EDGE_DIM;

    v8f acc[8];
#pragma unroll
    for (int t = 0; t < 8; ++t) acc[t] = (v8f){};

#pragma unroll
    for (int c = 0; c < 9; ++c) {          // K = 288 in 9 chunks of 32
        v16h a;
        if (c < 4)      a = load_a_frag_off(h, so, c * 32);
        else if (c < 8) a = load_a_frag_off(h, dofs, (c - 4) * 32);
        else            a = load_a_frag_off(edge_attr, eo, 0);
#pragma unroll
        for (int t = 0; t < 8; ++t)
            acc[t] = wmma_f16(a, load_b_frag_packed(w1p, c * 8 + t), acc[t]);
    }

    // hidden tile -> LDS with bias + relu
#pragma unroll
    for (int t = 0; t < 8; ++t) {
        const float bv = b1[t * 16 + ln];
        float* hrow = &hid[wave][hi ? 8 : 0][t * 16 + ln];   // + r*132
#pragma unroll
        for (int r = 0; r < 8; ++r)
            hrow[r * 132] = fmaxf(acc[t][r] + bv, 0.0f);
    }
    __syncthreads();

    // [16 x 128] @ [128 x 24]
    v8f o0 = {}, o1 = {};
#pragma unroll
    for (int c2 = 0; c2 < 4; ++c2) {
        v16h a2 = load_a_frag_lds(&hid[wave][ln][0], c2 * 32);
        o0 = wmma_f16(a2, load_b_frag_packed(w2p, c2 * 2 + 0), o0);
        o1 = wmma_f16(a2, load_b_frag_packed(w2p, c2 * 2 + 1), o1);
    }
#pragma unroll
    for (int t2 = 0; t2 < 2; ++t2) {
        const int col = t2 * 16 + ln;
        if (col < OUT) {
            const float bv = b2[col];
            const v8f o = t2 ? o1 : o0;
            const int ooff = (e0 + (hi ? 8 : 0)) * OUT + col;   // + r*24
#pragma unroll
            for (int r = 0; r < 8; ++r)
                out[ooff + r * OUT] = o[r] + bv;
        }
    }
}

// ---------------------------------------------------------------------------
extern "C" void kernel_launch(void* const* d_in, const int* in_sizes, int n_in,
                              void* d_out, int out_size, void* d_ws, size_t ws_size,
                              hipStream_t stream) {
    (void)in_sizes; (void)n_in; (void)out_size; (void)ws_size;

    const float* x         = (const float*)d_in[0];
    const int*   eidx      = (const int*)  d_in[1];
    const float* edge_attr = (const float*)d_in[2];
    const float* lin1_w    = (const float*)d_in[3];
    const float* lin1_b    = (const float*)d_in[4];
    const float* cls_w1    = (const float*)d_in[23];
    const float* cls_b1    = (const float*)d_in[24];
    const float* cls_w2    = (const float*)d_in[25];
    const float* cls_b2    = (const float*)d_in[26];

    const int* src = eidx;
    const int* dst = eidx + N_EDGES;

    float* h   = (float*)d_ws;
    float* agg = h   + (size_t)N_NODES * HID;
    float* t1  = agg + (size_t)N_NODES * HID;
    float* out = (float*)d_out;

    // f16 packed-weight area after the three f32 node buffers (~312 KB)
    _Float16* pk = (_Float16*)((char*)d_ws + (size_t)3 * N_NODES * HID * sizeof(float));
    _Float16* p_lin1 = pk;                       // (64/32)*8*512  = 8192
    _Float16* p_ew[3], *p_w1[3], *p_w2[3];
    _Float16* cur = p_lin1 + 8192;
    for (int L = 0; L < 3; ++L) {
        p_ew[L] = cur; cur += 1 * 8 * 512;       // 32x128
        p_w1[L] = cur; cur += 4 * 8 * 512;       // 128x128
        p_w2[L] = cur; cur += 4 * 8 * 512;       // 128x128
    }
    _Float16* p_cls1 = cur; cur += 9 * 8 * 512;  // 288x128
    _Float16* p_cls2 = cur;                      // 128x24 -> 128x32 padded

    auto pack = [&](const float* W, _Float16* P, int K, int N, int tiles) {
        const int total = (K / 32) * tiles * 512;
        pack_w_f16<<<(total + 255) / 256, 256, 0, stream>>>(W, P, K, N, tiles);
    };
    pack(lin1_w, p_lin1, NODE_DIM, HID, 8);
    for (int L = 0; L < 3; ++L) {
        pack((const float*)d_in[5 + 6 * L], p_ew[L], EDGE_DIM, HID, 8);
        pack((const float*)d_in[7 + 6 * L], p_w1[L], HID, HID, 8);
        pack((const float*)d_in[9 + 6 * L], p_w2[L], HID, HID, 8);
    }
    pack(cls_w1, p_cls1, 2 * HID + EDGE_DIM, HID, 8);
    pack(cls_w2, p_cls2, HID, OUT, 2);

    // h = x @ lin1_w + b ; dual-store seeds agg for layer 0 (z = h + agg)
    gemm_bias_act<false><<<N_NODES / 16, 256, 0, stream>>>(
        x, p_lin1, lin1_b, h, agg, N_NODES, NODE_DIM);

    for (int L = 0; L < 3; ++L) {
        const float* eb = (const float*)d_in[6  + 6 * L];
        const float* b1 = (const float*)d_in[8  + 6 * L];
        const float* b2 = (const float*)d_in[10 + 6 * L];

        const int waveTiles = N_EDGES / 16;      // 37500
        edge_msg_scatter<<<(waveTiles + 7) / 8, 256, 0, stream>>>(
            edge_attr, src, dst, p_ew[L], eb, h, agg, N_EDGES);

        gemm_bias_act<true><<<N_NODES / 16, 256, 0, stream>>>(
            agg, p_w1[L], b1, t1, nullptr, N_NODES, HID);
        // dual-store: h and next layer's agg seed (skip seed on last layer)
        gemm_bias_act<true><<<N_NODES / 16, 256, 0, stream>>>(
            t1, p_w2[L], b2, h, (L < 2) ? agg : nullptr, N_NODES, HID);
    }

    edge_classifier<<<N_EDGES / 64, 128, 0, stream>>>(
        h, src, dst, edge_attr, p_cls1, cls_b1, p_cls2, cls_b2, out, N_EDGES);
}
